// Qwen2MoeMLP_75960791597569
// MI455X (gfx1250) — compile-verified, matrix operand order checked by reading, not run on hardware
//
#include <hip/hip_runtime.h>
#include <math.h>

// ---------------------------------------------------------------------------
// Qwen2-MoE MLP for MI455X (gfx1250, wave32, WMMA bf16 16x16x32)
// T=2048 tokens, E=16 experts, H=2048 hidden, I=1408 intermediate, top-K=4
// ---------------------------------------------------------------------------

typedef __attribute__((ext_vector_type(16))) __bf16 v16bf;
typedef __attribute__((ext_vector_type(8)))  float  v8f;

#define T_TOK 2048
#define N_EXP 16
#define H_DIM 2048
#define I_DIM 1408
#define TOPK  4
#define TK    (T_TOK * TOPK)

// ---- workspace layout (bytes) ---------------------------------------------
#define OFF_CNT   0                       // 16 ints
#define OFF_OFFS  256                     // 17 ints
#define OFF_TOKE  1024                    // TK ints
#define OFF_TOKP  (OFF_TOKE + TK * 4)     // TK ints
#define OFF_TOKW  (OFF_TOKP + TK * 4)     // TK floats
#define OFF_BTOK  (OFF_TOKW + TK * 4)     // TK ints
#define OFF_BW    (OFF_BTOK + TK * 4)     // TK floats
#define OFF_XBF   (1 << 18)               // T*H bf16  (8 MB)
#define OFF_HBUF  (OFF_XBF + T_TOK * H_DIM * 2)  // TK*I bf16 (23 MB)

// K offset of the v-th packed pair in a 16-bit 16x32 A/B fragment.
// (ISA 7.12.2: lanes 0-15 hold K {0-7,16-23}, lanes 16-31 hold K {8-15,24-31})
__device__ __forceinline__ int kpair(int v, int half) {
  return ((v >> 2) << 4) | (half << 3) | ((v & 3) << 1);
}

__device__ __forceinline__ float fast_rcp(float x) {
#if __has_builtin(__builtin_amdgcn_rcpf)
  return __builtin_amdgcn_rcpf(x);   // v_rcp_f32 (1 TRANS op, ~1 ulp)
#else
  return 1.0f / x;
#endif
}

// ---------------------------------------------------------------------------
__global__ void zero_f4(float* p, int n) {
  int i = (blockIdx.x * blockDim.x + threadIdx.x) * 4;
  if (i < n) *(float4*)(p + i) = make_float4(0.f, 0.f, 0.f, 0.f);
}

__global__ void zero_cnt(int* cnt) {
  if (threadIdx.x < N_EXP) cnt[threadIdx.x] = 0;
}

__global__ void cvt_x_bf16(const float* __restrict__ x, __bf16* __restrict__ xbf) {
  int i = (blockIdx.x * blockDim.x + threadIdx.x) * 4;
  if (i >= T_TOK * H_DIM) return;
  float4 v = *(const float4*)(x + i);
  xbf[i + 0] = (__bf16)v.x;
  xbf[i + 1] = (__bf16)v.y;
  xbf[i + 2] = (__bf16)v.z;
  xbf[i + 3] = (__bf16)v.w;
}

// ---- router: softmax over 16 logits, top-4, renormalize, bucket by expert --
__global__ void router_topk(const float* __restrict__ logits, int* cnt,
                            int* tokE, int* tokP, float* tokW) {
  int t = blockIdx.x * blockDim.x + threadIdx.x;
  if (t >= T_TOK) return;
  float l[N_EXP];
  float mx = -3.4e38f;
#pragma unroll
  for (int e = 0; e < N_EXP; ++e) { l[e] = logits[t * N_EXP + e]; mx = fmaxf(mx, l[e]); }
#pragma unroll
  for (int e = 0; e < N_EXP; ++e) l[e] = __expf(l[e] - mx);

  unsigned taken = 0u;
  int   ids[TOPK];
  float vs[TOPK];
  float tsum = 0.f;
#pragma unroll
  for (int k = 0; k < TOPK; ++k) {
    float bv = -1.f; int bi = 0;
#pragma unroll
    for (int e = 0; e < N_EXP; ++e)
      if (!((taken >> e) & 1u) && l[e] > bv) { bv = l[e]; bi = e; }
    taken |= 1u << bi;
    ids[k] = bi; vs[k] = bv; tsum += bv;
  }
  // probs = l/Z; renormalized top-k weight = l[k] / sum(top-k l)  (Z cancels)
  float rts = fast_rcp(tsum);
#pragma unroll
  for (int k = 0; k < TOPK; ++k) {
    int e = ids[k];
    int pos = atomicAdd(&cnt[e], 1);
    tokE[t * TOPK + k] = e;
    tokP[t * TOPK + k] = pos;
    tokW[t * TOPK + k] = vs[k] * rts;
  }
}

__global__ void scan_offs(const int* __restrict__ cnt, int* offs) {
  if (threadIdx.x == 0 && blockIdx.x == 0) {
    int s = 0;
#pragma unroll
    for (int e = 0; e < N_EXP; ++e) { offs[e] = s; s += cnt[e]; }
    offs[N_EXP] = s;
  }
}

__global__ void scatter_slots(const int* __restrict__ tokE, const int* __restrict__ tokP,
                              const float* __restrict__ tokW, const int* __restrict__ offs,
                              int* btok, float* bwv) {
  int id = blockIdx.x * blockDim.x + threadIdx.x;
  if (id >= TK) return;
  int e = tokE[id];
  int slot = offs[e] + tokP[id];
  btok[slot] = id / TOPK;
  bwv[slot]  = tokW[id];
}

// ---- phase A: h[slot, i] = combine * silu(x·gateW) * (x·upW)  -------------
// block = 256 thr = 8 waves; wave tile = 32 tokens x 32 I-cols (2x2 WMMA tiles,
// gate + up accumulated simultaneously -> 8 WMMA per K=32 step)
__global__ __launch_bounds__(256)
void moe_gate_up(const float* __restrict__ gw, const float* __restrict__ uw,
                 const __bf16* __restrict__ xbf,
                 const int* __restrict__ cnt, const int* __restrict__ offs,
                 const int* __restrict__ btok, const float* __restrict__ bw,
                 __bf16* __restrict__ hbuf) {
  int e = blockIdx.z;
  int count = cnt[e];
  int mbase = blockIdx.y * 32;
  if (mbase >= count) return;
  int wave = threadIdx.x >> 5;
  int lane = threadIdx.x & 31;
  int nbase = blockIdx.x * 256 + wave * 32;
  if (nbase >= I_DIM) return;  // 1408 = 44 tiles of 32; tail waves idle
  int lo = lane & 15, half = lane >> 4;
  int off = offs[e];

  // A rows: lane 'lo' owns matrix row m (gathered token row, clamped pad)
  const __bf16* arow[2];
#pragma unroll
  for (int ms = 0; ms < 2; ++ms) {
    int m = mbase + ms * 16 + lo;
    int mc = m < count ? m : count - 1;
    arow[ms] = xbf + (size_t)btok[off + mc] * H_DIM;
  }
  // B rows: lane 'lo' owns matrix column n -> weight row i, contiguous in H(=K)
  const float* growp[2];
  const float* urowp[2];
#pragma unroll
  for (int ns = 0; ns < 2; ++ns) {
    int i = nbase + ns * 16 + lo;
    growp[ns] = gw + ((size_t)e * I_DIM + i) * H_DIM;
    urowp[ns] = uw + ((size_t)e * I_DIM + i) * H_DIM;
  }

  v8f accg[2][2] = {};
  v8f accu[2][2] = {};
  for (int k0 = 0; k0 < H_DIM; k0 += 32) {
    v16bf a[2], bg[2], bu[2];
#pragma unroll
    for (int ms = 0; ms < 2; ++ms)
#pragma unroll
      for (int v = 0; v < 8; ++v) {
        int k = k0 + kpair(v, half);
        a[ms][2 * v]     = arow[ms][k];
        a[ms][2 * v + 1] = arow[ms][k + 1];
      }
#pragma unroll
    for (int ns = 0; ns < 2; ++ns)
#pragma unroll
      for (int v = 0; v < 8; ++v) {
        int k = k0 + kpair(v, half);
        bg[ns][2 * v]     = (__bf16)growp[ns][k];
        bg[ns][2 * v + 1] = (__bf16)growp[ns][k + 1];
        bu[ns][2 * v]     = (__bf16)urowp[ns][k];
        bu[ns][2 * v + 1] = (__bf16)urowp[ns][k + 1];
      }
#pragma unroll
    for (int ms = 0; ms < 2; ++ms)
#pragma unroll
      for (int ns = 0; ns < 2; ++ns) {
        accg[ms][ns] = __builtin_amdgcn_wmma_f32_16x16x32_bf16(
            false, a[ms], false, bg[ns], (short)0, accg[ms][ns], false, false);
        accu[ms][ns] = __builtin_amdgcn_wmma_f32_16x16x32_bf16(
            false, a[ms], false, bu[ns], (short)0, accu[ms][ns], false, false);
      }
  }

  // C layout: lane holds column n=lo; VGPR r holds row m = r + 8*half.
#pragma unroll
  for (int ms = 0; ms < 2; ++ms)
#pragma unroll
    for (int r = 0; r < 8; ++r) {
      int m = mbase + ms * 16 + r + 8 * half;
      if (m >= count) continue;
      int slot = off + m;
      float cw = bw[slot];  // fold combine weight here (linear downstream)
#pragma unroll
      for (int ns = 0; ns < 2; ++ns) {
        float g = accg[ms][ns][r];
        float u = accu[ms][ns][r];
        // silu(g) = g * sigmoid(g); fast v_rcp instead of IEEE division
        float hv = g * fast_rcp(1.0f + __expf(-g)) * u * cw;
        hbuf[(size_t)slot * I_DIM + (nbase + ns * 16 + lo)] = (__bf16)hv;
      }
    }
}

// ---- phase B: out[tok, h] += h[slot,:] · downW[e,h,:]  --------------------
// wave tile = 32 tokens x 64 H-cols (2x4 WMMA tiles -> 8 WMMA per K=32 step;
// doubles reuse of the bf16 hbuf A fragments vs 2x2)
__global__ __launch_bounds__(256)
void moe_down(const float* __restrict__ dw, const __bf16* __restrict__ hbuf,
              const int* __restrict__ cnt, const int* __restrict__ offs,
              const int* __restrict__ btok, float* __restrict__ out) {
  int e = blockIdx.z;
  int count = cnt[e];
  int mbase = blockIdx.y * 32;
  if (mbase >= count) return;
  int wave = threadIdx.x >> 5;
  int lane = threadIdx.x & 31;
  int nbase = blockIdx.x * 512 + wave * 64;  // output H column base
  int lo = lane & 15, half = lane >> 4;
  int off = offs[e];

  const __bf16* arow[2];
#pragma unroll
  for (int ms = 0; ms < 2; ++ms) {
    int m = mbase + ms * 16 + lo;
    int mc = m < count ? m : count - 1;
    arow[ms] = hbuf + (size_t)(off + mc) * I_DIM;
  }
  const float* drow[4];
#pragma unroll
  for (int ns = 0; ns < 4; ++ns) {
    int hc = nbase + ns * 16 + lo;
    drow[ns] = dw + ((size_t)e * H_DIM + hc) * I_DIM;  // contiguous in I(=K)
  }

  v8f acc[2][4] = {};
  for (int k0 = 0; k0 < I_DIM; k0 += 32) {  // 1408 / 32 = 44 steps
    v16bf a[2], b[4];
#pragma unroll
    for (int ms = 0; ms < 2; ++ms)
#pragma unroll
      for (int v = 0; v < 8; ++v) {
        int k = k0 + kpair(v, half);
        a[ms][2 * v]     = arow[ms][k];
        a[ms][2 * v + 1] = arow[ms][k + 1];
      }
#pragma unroll
    for (int ns = 0; ns < 4; ++ns)
#pragma unroll
      for (int v = 0; v < 8; ++v) {
        int k = k0 + kpair(v, half);
        b[ns][2 * v]     = (__bf16)drow[ns][k];
        b[ns][2 * v + 1] = (__bf16)drow[ns][k + 1];
      }
#pragma unroll
    for (int ms = 0; ms < 2; ++ms)
#pragma unroll
      for (int ns = 0; ns < 4; ++ns)
        acc[ms][ns] = __builtin_amdgcn_wmma_f32_16x16x32_bf16(
            false, a[ms], false, b[ns], (short)0, acc[ms][ns], false, false);
  }

#pragma unroll
  for (int ms = 0; ms < 2; ++ms)
#pragma unroll
    for (int r = 0; r < 8; ++r) {
      int m = mbase + ms * 16 + r + 8 * half;
      if (m >= count) continue;
      int t = btok[off + m];
#pragma unroll
      for (int ns = 0; ns < 4; ++ns)
        atomicAdd(out + (size_t)t * H_DIM + (nbase + ns * 16 + lo), acc[ms][ns][r]);
    }
}

// ---------------------------------------------------------------------------
extern "C" void kernel_launch(void* const* d_in, const int* in_sizes, int n_in,
                              void* d_out, int out_size, void* d_ws, size_t ws_size,
                              hipStream_t stream) {
  const float* x  = (const float*)d_in[0];
  const float* rl = (const float*)d_in[1];
  const float* gw = (const float*)d_in[2];
  const float* uw = (const float*)d_in[3];
  const float* dw = (const float*)d_in[4];
  float* out = (float*)d_out;

  char* ws = (char*)d_ws;
  int*    cnt  = (int*)(ws + OFF_CNT);
  int*    offs = (int*)(ws + OFF_OFFS);
  int*    tokE = (int*)(ws + OFF_TOKE);
  int*    tokP = (int*)(ws + OFF_TOKP);
  float*  tokW = (float*)(ws + OFF_TOKW);
  int*    btok = (int*)(ws + OFF_BTOK);
  float*  bwv  = (float*)(ws + OFF_BW);
  __bf16* xbf  = (__bf16*)(ws + OFF_XBF);
  __bf16* hbuf = (__bf16*)(ws + OFF_HBUF);

  int nEl = T_TOK * H_DIM;
  zero_f4<<<(nEl / 4 + 255) / 256, 256, 0, stream>>>(out, nEl);
  zero_cnt<<<1, 32, 0, stream>>>(cnt);
  cvt_x_bf16<<<(nEl / 4 + 255) / 256, 256, 0, stream>>>(x, xbf);
  router_topk<<<(T_TOK + 255) / 256, 256, 0, stream>>>(rl, cnt, tokE, tokP, tokW);
  scan_offs<<<1, 32, 0, stream>>>(cnt, offs);
  scatter_slots<<<(TK + 255) / 256, 256, 0, stream>>>(tokE, tokP, tokW, offs, btok, bwv);

  dim3 ga((I_DIM + 255) / 256, T_TOK / 32, N_EXP);  // (6, 64, 16)
  moe_gate_up<<<ga, 256, 0, stream>>>(gw, uw, xbf, cnt, offs, btok, bwv, hbuf);

  dim3 gb(H_DIM / 512, T_TOK / 32, N_EXP);          // (4, 64, 16)
  moe_down<<<gb, 256, 0, stream>>>(dw, hbuf, cnt, offs, btok, out);
}
